// KVCacheAttention_31559419691621
// MI455X (gfx1250) — compile-verified, matrix operand order checked by reading, not run on hardware
//
#include <hip/hip_runtime.h>
#include <hip/hip_bf16.h>
#include <stdint.h>

// ---------------- model constants (from reference) ----------------
#define T_SEQ   2048
#define HID     4096
#define NH      32
#define NKV     8
#define DHEAD   128
#define GQA     (NH / NKV)          // 4
#define QW      (NH * DHEAD)        // 4096
#define KVW     (NKV * DHEAD)       // 1024
#define ROPE_BASE 500000.0f
#define SM_SCALE  0.08838834764831845f   // 1/sqrt(128)

typedef unsigned short u16;
typedef __attribute__((ext_vector_type(8)))  unsigned short v8u;
typedef __attribute__((ext_vector_type(16))) unsigned short v16u;
typedef __attribute__((ext_vector_type(16))) __bf16         v16bf;
typedef __attribute__((ext_vector_type(8)))  float          v8f;
typedef __attribute__((ext_vector_type(4)))  unsigned int   u32x4;
typedef __attribute__((ext_vector_type(4)))  int            i32x4;
typedef __attribute__((ext_vector_type(8)))  int            i32x8;
typedef int gv4i __attribute__((vector_size(16)));   // GCC-style int4 (builtin param type)

// ---- gfx1250 feature guards (compile-safe on both toolchains) ----
#if defined(__gfx1250__) && __has_builtin(__builtin_amdgcn_global_load_async_to_lds_b128) && __has_builtin(__builtin_amdgcn_s_wait_asynccnt)
#define USE_ASYNC_LDS 1
#else
#define USE_ASYNC_LDS 0
#endif

#if defined(__gfx1250__) && __has_builtin(__builtin_amdgcn_tensor_load_to_lds) && __has_builtin(__builtin_amdgcn_s_wait_tensorcnt)
#define USE_TDM 1
#else
#define USE_TDM 0
#endif

#define ASPACE(n) __attribute__((address_space(n)))
__device__ __forceinline__ ASPACE(3) gv4i* lds_v4(void* p) {
  // generic LDS pointer: low 32 bits are the LDS byte offset (ISA 10.2)
  return (ASPACE(3) gv4i*)(unsigned)(uintptr_t)p;
}
__device__ __forceinline__ ASPACE(1) gv4i* gbl_v4(const void* p) {
  return (ASPACE(1) gv4i*)(uintptr_t)p;
}

__device__ __forceinline__ u16 f2bf(float f) {
  union { float f; unsigned int u; } x; x.f = f;
  unsigned int r = x.u + 0x7fffu + ((x.u >> 16) & 1u);  // RNE
  return (u16)(r >> 16);
}
__device__ __forceinline__ float bf2f(u16 b) {
  union { unsigned int u; float f; } x; x.u = ((unsigned int)b) << 16;
  return x.f;
}
__device__ __forceinline__ v8f zero8() {
  v8f z;
#pragma unroll
  for (int i = 0; i < 8; ++i) z[i] = 0.0f;
  return z;
}

__device__ __forceinline__ v8f wmma_bf16(v16u a, v16u b, v8f c) {
  // D(f32 16x16) = A(bf16 16x32) * B(bf16 32x16) + C
  return __builtin_amdgcn_wmma_f32_16x16x32_bf16(
      false, __builtin_bit_cast(v16bf, a),
      false, __builtin_bit_cast(v16bf, b),
      (short)0, c, false, false);
}

// CDNA5 16-bit A/B fragment: lane holds one row/col (lane&15) of 32 K-values;
// lane>>4 selects the K-half interleave. `row` points at 32 contiguous u16.
__device__ __forceinline__ v16u load_frag(const u16* row, int kh) {
  v8u lo = *(const v8u*)(row + kh * 8);
  v8u hi = *(const v8u*)(row + 16 + kh * 8);
  v16u f;
#pragma unroll
  for (int i = 0; i < 8; ++i) { f[i] = lo[i]; f[i + 8] = hi[i]; }
  return f;
}

#if USE_TDM
// Issue one TDM 2D tile load: tile_d0 x tile_d1 elements of 2 bytes from a
// row-major tensor with row stride `stride_elems`, into LDS at `lds_dst`.
__device__ __forceinline__ void tdm_load_tile_2d(const u16* gsrc, void* lds_dst,
                                                 int tile_d0, int tile_d1,
                                                 int tensor_d0, int tensor_d1,
                                                 int stride_elems) {
  unsigned long long ga = (unsigned long long)(uintptr_t)gsrc;  // byte addr
  unsigned lds_off = (unsigned)(uintptr_t)lds_dst;
  u32x4 g0;
  g0[0] = 1u;                                   // count=1, user-mode D#
  g0[1] = lds_off;                              // lds_addr
  g0[2] = (unsigned)(ga & 0xffffffffu);         // global_addr[31:0]
  g0[3] = (unsigned)((ga >> 32) & 0x01ffffffu)  // global_addr[56:32]
          | 0x80000000u;                        // type=2 ("image")
  i32x8 g1;
  g1[0] = (int)(1u << 16);                      // data_size=1 -> 2 bytes
  g1[1] = (int)(((unsigned)tensor_d0 & 0xffffu) << 16);        // dim0[15:0]
  g1[2] = (int)((((unsigned)tensor_d0 >> 16) & 0xffffu)        // dim0[31:16]
          | (((unsigned)tensor_d1 & 0xffffu) << 16));          // dim1[15:0]
  g1[3] = (int)((((unsigned)tensor_d1 >> 16) & 0xffffu)        // dim1[31:16]
          | (((unsigned)tile_d0 & 0xffffu) << 16));            // tile_dim0
  g1[4] = (int)((unsigned)tile_d1 & 0xffffu);                  // tile_dim1 (tile_dim2=0)
  g1[5] = (int)stride_elems;                                   // dim0_stride[31:0]
  g1[6] = 0;                                                   // stride hi / dim1_stride lo
  g1[7] = 0;
  i32x4 z4 = {0, 0, 0, 0};
#if __clang_major__ >= 23
  i32x8 z8 = {0, 0, 0, 0, 0, 0, 0, 0};
  __builtin_amdgcn_tensor_load_to_lds(g0, g1, z4, z4, z8, 0);
#else
  __builtin_amdgcn_tensor_load_to_lds(g0, g1, z4, z4, 0);
#endif
}
#endif

// ---------------- fp32 -> bf16 conversion ----------------
__global__ void cvt_f32_to_bf16(const float* __restrict__ in,
                                u16* __restrict__ out, int n) {
  int i = blockIdx.x * blockDim.x + threadIdx.x;
  if (i < n) out[i] = f2bf(in[i]);
}

// ---------------- bf16 GEMM: C[M,N] = A[M,K] @ B[K,N] ----------------
// 128x128x32 block tile, 256 threads = 8 wave32 in a 4(M) x 2(N) grid,
// each wave: 2x4 = 8 WMMA 16x16 accumulators.
template <bool F32OUT>
__global__ __launch_bounds__(256)
void gemm_bf16_wmma(const u16* __restrict__ A, const u16* __restrict__ B,
                    void* __restrict__ Cv, int M, int N, int K) {
  constexpr int BM = 128, BN = 128, BK = 32;
  __shared__ __align__(16) u16 As[BM][BK];       // row-major A tile
  __shared__ __align__(16) u16 Bt[BN][BK + 8];   // B tile transposed (n-major)

  const int tid  = threadIdx.x;
  const int lane = tid & 31;
  const int wave = tid >> 5;
  const int lr   = lane & 15;
  const int kh   = lane >> 4;
  const int wm   = wave & 3;   // 32-row strip
  const int wn   = wave >> 2;  // 64-col strip
  const int m0   = blockIdx.y * BM;
  const int n0   = blockIdx.x * BN;

  v8f acc[2][4];
#pragma unroll
  for (int i = 0; i < 2; ++i)
#pragma unroll
    for (int j = 0; j < 4; ++j) acc[i][j] = zero8();

  for (int k0 = 0; k0 < K; k0 += BK) {
    __syncthreads();
    // ---- A tile 128x32 ----
#if USE_TDM
    if (wave == 0) {
      // 2D tile: 32 contiguous K elems x 128 rows, row stride K
      tdm_load_tile_2d(A + (size_t)m0 * K + k0, &As[0][0],
                       /*tile*/ BK, BM, /*tensor*/ K - k0, M - m0, K);
      __builtin_amdgcn_s_wait_tensorcnt(0);
    }
#elif USE_ASYNC_LDS
    for (int i = tid; i < BM * (BK / 8); i += 256) {
      int m = i >> 2, kv = i & 3;
      __builtin_amdgcn_global_load_async_to_lds_b128(
          gbl_v4(A + (size_t)(m0 + m) * K + k0 + kv * 8),
          lds_v4(&As[m][kv * 8]), 0, 0);
    }
#else
    for (int i = tid; i < BM * (BK / 8); i += 256) {
      int m = i >> 2, kv = i & 3;
      *(v8u*)&As[m][kv * 8] =
          *(const v8u*)(A + (size_t)(m0 + m) * K + k0 + kv * 8);
    }
#endif
    // ---- B tile: read rows of B, scatter transposed into Bt[n][k] ----
    for (int i = tid; i < (BN / 8) * BK; i += 256) {
      int kk = i >> 4, nv = i & 15;
      v8u val = *(const v8u*)(B + (size_t)(k0 + kk) * N + n0 + nv * 8);
#pragma unroll
      for (int j = 0; j < 8; ++j) Bt[nv * 8 + j][kk] = val[j];
    }
    if (k0 + BK < K) {  // gfx1250 global_prefetch_b8 of next tiles
      __builtin_prefetch(A + (size_t)(m0 + (tid & 127)) * K + k0 + BK, 0, 0);
      __builtin_prefetch(B + (size_t)(k0 + BK + (tid & 31)) * N + n0, 0, 0);
    }
#if USE_ASYNC_LDS && !USE_TDM
    __builtin_amdgcn_s_wait_asynccnt(0);
#endif
    __syncthreads();

    v16u afrag[2], bfrag[4];
#pragma unroll
    for (int i = 0; i < 2; ++i)
      afrag[i] = load_frag(&As[wm * 32 + i * 16 + lr][0], kh);
#pragma unroll
    for (int j = 0; j < 4; ++j)
      bfrag[j] = load_frag(&Bt[wn * 64 + j * 16 + lr][0], kh);
#pragma unroll
    for (int i = 0; i < 2; ++i)
#pragma unroll
      for (int j = 0; j < 4; ++j)
        acc[i][j] = wmma_bf16(afrag[i], bfrag[j], acc[i][j]);
  }

  // C layout: VGPR r -> M = r + 8*(lane/16), N = lane&15
#pragma unroll
  for (int i = 0; i < 2; ++i)
#pragma unroll
    for (int j = 0; j < 4; ++j) {
      int col = n0 + wn * 64 + j * 16 + lr;
#pragma unroll
      for (int r = 0; r < 8; ++r) {
        int row = m0 + wm * 32 + i * 16 + r + 8 * kh;
        if constexpr (F32OUT)
          ((float*)Cv)[(size_t)row * N + col] = acc[i][j][r];
        else
          ((u16*)Cv)[(size_t)row * N + col] = f2bf(acc[i][j][r]);
      }
    }
}

// ---------------- RoPE (in place, bf16 buffer [T, n_heads*128]) ----------------
__global__ void rope_bf16(u16* __restrict__ buf, int n_heads,
                          const int* __restrict__ pos_p) {
  const int half = DHEAD / 2;
  int idx = blockIdx.x * blockDim.x + threadIdx.x;
  int total = T_SEQ * n_heads * half;
  if (idx >= total) return;
  int d = idx % half;
  int h = (idx / half) % n_heads;
  int t = idx / (half * n_heads);
  float inv = __powf(ROPE_BASE, -(float)d / (float)half);
  float ang = (float)(pos_p[0] + t) * inv;
  float s, c;
  __sincosf(ang, &s, &c);
  u16* row = buf + (size_t)t * (n_heads * DHEAD) + h * DHEAD;
  float x1 = bf2f(row[d]);
  float x2 = bf2f(row[half + d]);
  row[d]        = f2bf(x1 * c - x2 * s);
  row[half + d] = f2bf(x2 * c + x1 * s);
}

// ---------------- Flash attention (causal, GQA 4:1), all-WMMA ----------------
// grid = (NH, T/128). 256 threads = 8 waves, each wave owns 16 q rows.
__global__ __launch_bounds__(256)
void flash_attn_wmma(const u16* __restrict__ Q, const u16* __restrict__ Kc,
                     const u16* __restrict__ Vc, u16* __restrict__ O,
                     const int* __restrict__ pos_p) {
  constexpr int QT = 128;  // q rows per workgroup
  constexpr int KT = 64;   // keys per inner tile
  __shared__ __align__(16) u16 Ks[KT][DHEAD + 8];   // [key][d]   (B for Q*K^T)
  __shared__ __align__(16) u16 Vt[DHEAD][KT + 8];   // [d][key]   (B for P*V)
  __shared__ __align__(16) u16 Ps[QT][KT + 8];      // [qrow][key] (A for P*V)

  const int h   = blockIdx.x;
  const int qt  = blockIdx.y;
  const int hk  = h / GQA;
  const int tid = threadIdx.x;
  const int lane = tid & 31;
  const int wave = tid >> 5;
  const int lr = lane & 15;
  const int kh = lane >> 4;
  const int pos = pos_p[0];

  // Q fragments for this wave's 16 rows: 4 K-steps of 32 over DH=128
  v16u qfrag[4];
  {
    int qrow = qt * QT + wave * 16 + lr;
    const u16* qb = Q + (size_t)qrow * QW + h * DHEAD;
#pragma unroll
    for (int ks = 0; ks < 4; ++ks) qfrag[ks] = load_frag(qb + ks * 32, kh);
  }

  v8f osum[8];  // 16 x 128 output strip (8 N-tiles)
#pragma unroll
  for (int dt = 0; dt < 8; ++dt) osum[dt] = zero8();
  float mrow[8], lrow[8];
#pragma unroll
  for (int r = 0; r < 8; ++r) { mrow[r] = -1e30f; lrow[r] = 0.0f; }

  const int last_key = pos + qt * QT + (QT - 1);
  const int n_kt = last_key / KT + 1;

  for (int kt = 0; kt < n_kt; ++kt) {
    __syncthreads();
    // ---- stage K tile row-major [key][d] ----
#if USE_ASYNC_LDS
    for (int i = tid; i < KT * (DHEAD / 8); i += 256) {
      int key = i >> 4, dv = i & 15;
      __builtin_amdgcn_global_load_async_to_lds_b128(
          gbl_v4(Kc + (size_t)(kt * KT + key) * KVW + hk * DHEAD + dv * 8),
          lds_v4(&Ks[key][dv * 8]), 0, 0);
    }
#else
    for (int i = tid; i < KT * (DHEAD / 8); i += 256) {
      int key = i >> 4, dv = i & 15;
      *(v8u*)&Ks[key][dv * 8] = *(const v8u*)(
          Kc + (size_t)(kt * KT + key) * KVW + hk * DHEAD + dv * 8);
    }
#endif
    // ---- stage V tile transposed [d][key] (manual: needs transpose) ----
    for (int i = tid; i < KT * (DHEAD / 8); i += 256) {
      int key = i >> 4, dv = i & 15;
      v8u val = *(const v8u*)(
          Vc + (size_t)(kt * KT + key) * KVW + hk * DHEAD + dv * 8);
#pragma unroll
      for (int j = 0; j < 8; ++j) Vt[dv * 8 + j][key] = val[j];
    }
    if (kt + 1 < n_kt) {
      __builtin_prefetch(
          Kc + (size_t)((kt + 1) * KT + (tid & 63)) * KVW + hk * DHEAD, 0, 0);
      __builtin_prefetch(
          Vc + (size_t)((kt + 1) * KT + (tid & 63)) * KVW + hk * DHEAD, 0, 0);
    }
#if USE_ASYNC_LDS
    __builtin_amdgcn_s_wait_asynccnt(0);
#endif
    __syncthreads();

    // S = Q * K^T  (16 x 64 per wave): 16 WMMAs
    v8f sacc[4];
#pragma unroll
    for (int nt = 0; nt < 4; ++nt) sacc[nt] = zero8();
#pragma unroll
    for (int ks = 0; ks < 4; ++ks) {
#pragma unroll
      for (int nt = 0; nt < 4; ++nt) {
        v16u b = load_frag(&Ks[nt * 16 + lr][ks * 32], kh);
        sacc[nt] = wmma_bf16(qfrag[ks], b, sacc[nt]);
      }
    }

    // causal mask + scale (C layout: M = r + 8*kh, N = lr)
#pragma unroll
    for (int nt = 0; nt < 4; ++nt) {
#pragma unroll
      for (int r = 0; r < 8; ++r) {
        int qabs = pos + qt * QT + wave * 16 + r + 8 * kh;
        int kabs = kt * KT + nt * 16 + lr;
        float sv = sacc[nt][r] * SM_SCALE;
        sacc[nt][r] = (kabs <= qabs) ? sv : -1e30f;
      }
    }

    // online softmax: reductions over the 16-lane N group via shfl_xor
#pragma unroll
    for (int r = 0; r < 8; ++r) {
      float m = sacc[0][r];
#pragma unroll
      for (int nt = 1; nt < 4; ++nt) m = fmaxf(m, sacc[nt][r]);
#pragma unroll
      for (int off = 1; off < 16; off <<= 1)
        m = fmaxf(m, __shfl_xor(m, off, 32));
      float mnew  = fmaxf(mrow[r], m);
      float alpha = __expf(mrow[r] - mnew);
      mrow[r] = mnew;
      lrow[r] *= alpha;
#pragma unroll
      for (int dt = 0; dt < 8; ++dt) osum[dt][r] *= alpha;
      float rs = 0.0f;
#pragma unroll
      for (int nt = 0; nt < 4; ++nt) {
        float p = __expf(sacc[nt][r] - mnew);
        sacc[nt][r] = p;
        rs += p;
      }
#pragma unroll
      for (int off = 1; off < 16; off <<= 1)
        rs += __shfl_xor(rs, off, 32);
      lrow[r] += rs;
    }

    // P -> LDS as bf16 in A-matrix-friendly row-major layout
#pragma unroll
    for (int nt = 0; nt < 4; ++nt)
#pragma unroll
      for (int r = 0; r < 8; ++r)
        Ps[wave * 16 + r + 8 * kh][nt * 16 + lr] = f2bf(sacc[nt][r]);
    __syncthreads();

    // O += P * V  (16x64 @ 64x128): 16 WMMAs
#pragma unroll
    for (int ks = 0; ks < 2; ++ks) {
      v16u a = load_frag(&Ps[wave * 16 + lr][ks * 32], kh);
#pragma unroll
      for (int dt = 0; dt < 8; ++dt) {
        v16u b = load_frag(&Vt[dt * 16 + lr][ks * 32], kh);
        osum[dt] = wmma_bf16(a, b, osum[dt]);
      }
    }
  }

  // normalize and store attn (bf16, layout [T, H*DH])
#pragma unroll
  for (int r = 0; r < 8; ++r) {
    float invl = 1.0f / lrow[r];
    int trow = qt * QT + wave * 16 + r + 8 * kh;
#pragma unroll
    for (int dt = 0; dt < 8; ++dt)
      O[(size_t)trow * QW + h * DHEAD + dt * 16 + lr] =
          f2bf(osum[dt][r] * invl);
  }
}

// ---------------- host-side orchestration ----------------
extern "C" void kernel_launch(void* const* d_in, const int* in_sizes, int n_in,
                              void* d_out, int out_size, void* d_ws, size_t ws_size,
                              hipStream_t stream) {
  (void)in_sizes; (void)n_in; (void)out_size; (void)ws_size;
  const float* hs = (const float*)d_in[0];
  const float* Wq = (const float*)d_in[1];
  const float* Wk = (const float*)d_in[2];
  const float* Wv = (const float*)d_in[3];
  const float* Wo = (const float*)d_in[4];
  const int*  pos = (const int*)d_in[5];
  float* out = (float*)d_out;

  char* ws = (char*)d_ws;
  size_t off = 0;
  auto take = [&](size_t bytes) -> char* {
    char* p = ws + off;
    off += (bytes + 255) & ~(size_t)255;
    return p;
  };
  u16* hsb = (u16*)take((size_t)T_SEQ * HID * 2);   // hidden bf16
  u16* wqb = (u16*)take((size_t)HID * QW * 2);
  u16* wkb = (u16*)take((size_t)HID * KVW * 2);
  u16* wvb = (u16*)take((size_t)HID * KVW * 2);
  u16* wob = (u16*)take((size_t)QW * HID * 2);
  u16* qb  = (u16*)take((size_t)T_SEQ * QW * 2);
  u16* kb  = (u16*)take((size_t)T_SEQ * KVW * 2);
  u16* vb  = (u16*)take((size_t)T_SEQ * KVW * 2);
  u16* ab  = (u16*)take((size_t)T_SEQ * QW * 2);    // attention output bf16

  dim3 blk(256);
  auto cvt = [&](const float* src, u16* dst, size_t n) {
    cvt_f32_to_bf16<<<(unsigned)((n + 255) / 256), blk, 0, stream>>>(src, dst, (int)n);
  };
  cvt(hs, hsb, (size_t)T_SEQ * HID);
  cvt(Wq, wqb, (size_t)HID * QW);
  cvt(Wk, wkb, (size_t)HID * KVW);
  cvt(Wv, wvb, (size_t)HID * KVW);
  cvt(Wo, wob, (size_t)QW * HID);

  // QKV projections (bf16 out)
  gemm_bf16_wmma<false><<<dim3(QW / 128,  T_SEQ / 128), blk, 0, stream>>>(
      hsb, wqb, qb, T_SEQ, QW, HID);
  gemm_bf16_wmma<false><<<dim3(KVW / 128, T_SEQ / 128), blk, 0, stream>>>(
      hsb, wkb, kb, T_SEQ, KVW, HID);
  gemm_bf16_wmma<false><<<dim3(KVW / 128, T_SEQ / 128), blk, 0, stream>>>(
      hsb, wvb, vb, T_SEQ, KVW, HID);

  // RoPE on q and k
  rope_bf16<<<(T_SEQ * NH  * (DHEAD / 2) + 255) / 256, blk, 0, stream>>>(qb, NH,  pos);
  rope_bf16<<<(T_SEQ * NKV * (DHEAD / 2) + 255) / 256, blk, 0, stream>>>(kb, NKV, pos);

  // causal GQA flash attention
  flash_attn_wmma<<<dim3(NH, T_SEQ / 128), blk, 0, stream>>>(qb, kb, vb, ab, pos);

  // output projection (fp32 out)
  gemm_bf16_wmma<true><<<dim3(HID / 128, T_SEQ / 128), blk, 0, stream>>>(
      ab, wob, out, T_SEQ, HID, QW);
}
